// EquivariantMetaLayer_65266323030018
// MI455X (gfx1250) — matrix operand order repaired and checked by reading, not compiled, not run on hardware
//
#include <hip/hip_runtime.h>
#include <hip/hip_bf16.h>
#include <cstdint>
#include <cstddef>

typedef __attribute__((ext_vector_type(16))) __bf16 v16bf;
typedef __attribute__((ext_vector_type(8)))  __bf16 v8bf;
typedef __attribute__((ext_vector_type(8)))  float  v8f;

#define NNODE 50000
#define NEDGE 1600000
#define NB    8
#define UD    64
#define HID   128
#define NAW   70       /* node_attr width = 2*ND + SCAL */
#define EDGED 16
#define K1    160      /* e_in 147 padded to mult of 32 */
#define K2    128
#define K3    192      /* n1_in = 64 + 128 */
#define N2IN  134      /* n2_in = 64 + 6 + 64 */

/* ---- edge-kernel LDS layout (bytes) ---- */
#define OFF_W1T 0
#define OFF_W2T (OFF_W1T + HID*K1*2)            /* 40960 */
#define OFF_W3T (OFF_W2T + HID*K2*2)            /* 73728 */
#define OFF_B1  (OFF_W3T + HID*K3*2)            /* 122880 */
#define OFF_B2  (OFF_B1 + 512)
#define OFF_B3  (OFF_B2 + 512)
#define OFF_W1B (OFF_B3 + 512)
#define OFF_STAGE (OFF_W1B + 512)               /* 124928 */
#define STAGE_PER_WAVE (32*K3*2 + 32*K2*2)      /* X(32x192) + Y(32x128) = 20480 */
#define EDGE_SMEM (OFF_STAGE + 8*STAGE_PER_WAVE)/* 288768 */
#define NODE_SMEM ((N2IN*HID + HID + HID*3 + 128*136)*4) /* 140288 */

__device__ __forceinline__ void lds_fence() {
  asm volatile("s_wait_dscnt 0" ::: "memory");
}
/* silu with fast hw reciprocal (v_rcp_f32) instead of IEEE divide */
__device__ __forceinline__ float silu(float x) {
  return x * __builtin_amdgcn_rcpf(1.0f + __expf(-x));
}

__device__ __forceinline__ uint32_t packbf2(float a, float b) {
  union { __bf16 h[2]; uint32_t u; } z;
  z.h[0] = (__bf16)a; z.h[1] = (__bf16)b; return z.u;
}

/* A-matrix 16x32 bf16 fragment: lane half h reads K runs {kb+8h..+7} and {kb+16+8h..+7} (ISA 7.12.2) */
__device__ __forceinline__ v16bf fragA(const __bf16* row, int kbase, int h) {
  v8bf lo = *(const v8bf*)(row + kbase + 8*h);
  v8bf hi = *(const v8bf*)(row + kbase + 16 + 8*h);
  v16bf r;
#pragma unroll
  for (int i = 0; i < 8; ++i) { r[i] = lo[i]; r[i+8] = hi[i]; }
  return r;
}
/* B-matrix 32x16 bf16 fragment from transposed weights Wt[n][k]: lane half h reads K runs {kb+8h..+7} and {kb+16+8h..+7} */
__device__ __forceinline__ v16bf fragB(const __bf16* row, int kbase, int h) {
  v8bf lo = *(const v8bf*)(row + kbase + 8*h);
  v8bf hi = *(const v8bf*)(row + kbase + 16 + 8*h);
  v16bf r;
#pragma unroll
  for (int i = 0; i < 8; ++i) { r[i] = lo[i]; r[i+8] = hi[i]; }
  return r;
}

/* ------------------------------------------------------------------ */
/* Kernel 1: per-edge MLP chain via bf16 WMMA.                         */
/* Each wave owns 32 edges (2 M-tiles) per iteration so every weight   */
/* B-fragment LDS load feeds two WMMAs (halves LDS traffic per FLOP).  */
/* ------------------------------------------------------------------ */
__global__ void __launch_bounds__(256)
edge_kernel(const float* __restrict__ node_attr,
            const int*   __restrict__ eidx,
            const float* __restrict__ edge_attr,
            const float* __restrict__ We1, const float* __restrict__ be1,
            const float* __restrict__ We2, const float* __restrict__ be2,
            const float* __restrict__ Wn1a, const float* __restrict__ bn1a,
            const float* __restrict__ Wn1b, const float* __restrict__ bn1b,
            float* __restrict__ ef_out,
            float* __restrict__ seg_vec,   /* N*3 */
            float* __restrict__ seg_cnt)   /* N   */
{
  extern __shared__ __align__(16) char smem[];
  __bf16* W1t  = (__bf16*)(smem + OFF_W1T);
  __bf16* W2t  = (__bf16*)(smem + OFF_W2T);
  __bf16* W3t  = (__bf16*)(smem + OFF_W3T);
  float*  b1s  = (float*)(smem + OFF_B1);
  float*  b2s  = (float*)(smem + OFF_B2);
  float*  b3s  = (float*)(smem + OFF_B3);
  float*  w1bs = (float*)(smem + OFF_W1B);
  const int tid = threadIdx.x;

  /* preload weights, transposed [n][k], fp32 -> bf16 */
  for (int i = tid; i < HID*K1; i += 256) W1t[i] = (__bf16)0.0f;
  __syncthreads();
  for (int i = tid; i < 147*HID; i += 256) { int k = i >> 7, n = i & 127; W1t[n*K1 + k] = (__bf16)We1[i]; }
  for (int i = tid; i < HID*HID; i += 256) { int k = i >> 7, n = i & 127; W2t[n*K2 + k] = (__bf16)We2[i]; }
  for (int i = tid; i < K3*HID;  i += 256) { int k = i >> 7, n = i & 127; W3t[n*K3 + k] = (__bf16)Wn1a[i]; }
  if (tid < 128) { b1s[tid] = be1[tid]; b2s[tid] = be2[tid]; b3s[tid] = bn1a[tid]; w1bs[tid] = Wn1b[tid]; }
  __syncthreads();

  const float bn1b0 = bn1b[0];
  const int wave = tid >> 5, lane = tid & 31;
  const int h = lane >> 4, mr = lane & 15;
  __bf16* Xw = (__bf16*)(smem + OFF_STAGE + wave*STAGE_PER_WAVE);   /* 32 x K3 */
  __bf16* Yw = Xw + 32*K3;                                          /* 32 x K2 */
  const int ntiles = NEDGE / 32;

  for (int tile = blockIdx.x*8 + wave; tile < ntiles; tile += gridDim.x*8) {
    const int e0 = tile*32;

    int nexttile = tile + gridDim.x*8;
    if (nexttile < ntiles) {
      __builtin_prefetch(&eidx[nexttile*32 + mr], 0, 0);
      __builtin_prefetch(&eidx[NEDGE + nexttile*32 + mr], 0, 0);
    }

    /* ---- Phase A: build X[m][0:160] = [scal[row] | scal[col] | edge_attr | radial | pad] ---- */
#pragma unroll
    for (int mt = 0; mt < 2; ++mt) {
      const int e = e0 + mt*16 + mr;
      const int r = eidx[e], c = eidx[NEDGE + e];
      __bf16* Xm = Xw + (mt*16 + mr)*K3;
      if (h == 0) {
        const float* sp = node_attr + (size_t)r*NAW + 6;
#pragma unroll
        for (int j = 0; j < 64; j += 2) {
          float2 f = *(const float2*)(sp + j);
          *(uint32_t*)(Xm + j) = packbf2(f.x, f.y);
        }
        const float* ea = edge_attr + (size_t)e*EDGED;
#pragma unroll
        for (int j = 0; j < 16; j += 2) {
          float2 f = *(const float2*)(ea + j);
          *(uint32_t*)(Xm + 128 + j) = packbf2(f.x, f.y);
        }
      } else {
        const float* sp = node_attr + (size_t)c*NAW + 6;
#pragma unroll
        for (int j = 0; j < 64; j += 2) {
          float2 f = *(const float2*)(sp + j);
          *(uint32_t*)(Xm + 64 + j) = packbf2(f.x, f.y);
        }
        const float* nr = node_attr + (size_t)r*NAW;
        const float* nc = node_attr + (size_t)c*NAW;
        float pd0 = nr[0]-nc[0], pd1 = nr[1]-nc[1], pd2 = nr[2]-nc[2];
        float vd0 = nr[3]-nc[3], vd1 = nr[4]-nc[4], vd2 = nr[5]-nc[5];
        float fx  = sqrtf(pd0*pd0 + pd1*pd1 + pd2*pd2);
        float fv  = sqrtf(vd0*vd0 + vd1*vd1 + vd2*vd2);
        float fxv = pd0*vd0 + pd1*vd1 + pd2*vd2;
        Xm[144] = (__bf16)fx; Xm[145] = (__bf16)fv; Xm[146] = (__bf16)fxv;
#pragma unroll
        for (int j = 147; j < 160; ++j) Xm[j] = (__bf16)0.0f;
      }
    }
    lds_fence();

    /* ---- Layer 1: silu(X @ We1 + be1) -> Yw (bf16) ---- */
    v16bf a1[2][5];
#pragma unroll
    for (int mt = 0; mt < 2; ++mt)
#pragma unroll
      for (int kc = 0; kc < 5; ++kc)
        a1[mt][kc] = fragA(Xw + (mt*16 + mr)*K3, kc*32, h);
#pragma unroll
    for (int nt = 0; nt < 8; ++nt) {
      const int n = nt*16 + mr;
      float bias = b1s[n];
      v8f acc0, acc1;
#pragma unroll
      for (int v = 0; v < 8; ++v) { acc0[v] = bias; acc1[v] = bias; }
      const __bf16* Brow = W1t + n*K1;
#pragma unroll
      for (int kc = 0; kc < 5; ++kc) {
        v16bf b = fragB(Brow, kc*32, h);
        acc0 = __builtin_amdgcn_wmma_f32_16x16x32_bf16(false, a1[0][kc], false, b, (short)0, acc0, false, false);
        acc1 = __builtin_amdgcn_wmma_f32_16x16x32_bf16(false, a1[1][kc], false, b, (short)0, acc1, false, false);
      }
#pragma unroll
      for (int v = 0; v < 8; ++v) {
        int mm = v + 8*h;
        Yw[mm*K2 + n]        = (__bf16)silu(acc0[v]);
        Yw[(16+mm)*K2 + n]   = (__bf16)silu(acc1[v]);
      }
    }
    lds_fence();

    /* ---- Layer 2: ef = Yw @ We2 + be2 -> global + Xw[:,64:192] ---- */
    float* efo = ef_out + (size_t)e0 * HID;
    v16bf a2[2][4];
#pragma unroll
    for (int mt = 0; mt < 2; ++mt)
#pragma unroll
      for (int kc = 0; kc < 4; ++kc)
        a2[mt][kc] = fragA(Yw + (mt*16 + mr)*K2, kc*32, h);
#pragma unroll
    for (int nt = 0; nt < 8; ++nt) {
      const int n = nt*16 + mr;
      float bias = b2s[n];
      v8f acc0, acc1;
#pragma unroll
      for (int v = 0; v < 8; ++v) { acc0[v] = bias; acc1[v] = bias; }
      const __bf16* Brow = W2t + n*K2;
#pragma unroll
      for (int kc = 0; kc < 4; ++kc) {
        v16bf b = fragB(Brow, kc*32, h);
        acc0 = __builtin_amdgcn_wmma_f32_16x16x32_bf16(false, a2[0][kc], false, b, (short)0, acc0, false, false);
        acc1 = __builtin_amdgcn_wmma_f32_16x16x32_bf16(false, a2[1][kc], false, b, (short)0, acc1, false, false);
      }
#pragma unroll
      for (int v = 0; v < 8; ++v) {
        int mm = v + 8*h;
        float x0 = acc0[v], x1 = acc1[v];
        efo[(size_t)mm*HID + n]        = x0;         /* fp32 output */
        efo[(size_t)(16+mm)*HID + n]   = x1;
        Xw[mm*K3 + 64 + n]             = (__bf16)x0; /* layer-3 input */
        Xw[(16+mm)*K3 + 64 + n]        = (__bf16)x1;
      }
    }
    lds_fence();

    /* ---- Layer 3: silu([scal_row|ef] @ Wn1a + bn1a) . Wn1b + bn1b ---- */
    /* K-chunks 0,1 (= scal[row], X[0:64]) unchanged since layer 1: reuse a1[mt][0..1] */
    v16bf a3[2][6];
#pragma unroll
    for (int mt = 0; mt < 2; ++mt) {
      a3[mt][0] = a1[mt][0];
      a3[mt][1] = a1[mt][1];
#pragma unroll
      for (int kc = 2; kc < 6; ++kc)
        a3[mt][kc] = fragA(Xw + (mt*16 + mr)*K3, kc*32, h);
    }
    float ps[2][8];
#pragma unroll
    for (int mt = 0; mt < 2; ++mt)
#pragma unroll
      for (int v = 0; v < 8; ++v) ps[mt][v] = 0.0f;
#pragma unroll
    for (int nt = 0; nt < 8; ++nt) {
      const int n = nt*16 + mr;
      float bias = b3s[n];
      v8f acc0, acc1;
#pragma unroll
      for (int v = 0; v < 8; ++v) { acc0[v] = bias; acc1[v] = bias; }
      const __bf16* Brow = W3t + n*K3;
#pragma unroll
      for (int kc = 0; kc < 6; ++kc) {
        v16bf b = fragB(Brow, kc*32, h);
        acc0 = __builtin_amdgcn_wmma_f32_16x16x32_bf16(false, a3[0][kc], false, b, (short)0, acc0, false, false);
        acc1 = __builtin_amdgcn_wmma_f32_16x16x32_bf16(false, a3[1][kc], false, b, (short)0, acc1, false, false);
      }
      float w = w1bs[n];
#pragma unroll
      for (int v = 0; v < 8; ++v) {
        ps[0][v] += silu(acc0[v]) * w;
        ps[1][v] += silu(acc1[v]) * w;
      }
    }
    /* reduce over n (16 lanes per half), then scatter segment sums */
#pragma unroll
    for (int mt = 0; mt < 2; ++mt) {
#pragma unroll
      for (int v = 0; v < 8; ++v) {
        ps[mt][v] += __shfl_xor(ps[mt][v], 1, 16);
        ps[mt][v] += __shfl_xor(ps[mt][v], 2, 16);
        ps[mt][v] += __shfl_xor(ps[mt][v], 4, 16);
        ps[mt][v] += __shfl_xor(ps[mt][v], 8, 16);
      }
      float sc = ps[mt][0];
#pragma unroll
      for (int v = 1; v < 8; ++v) if (mr == v) sc = ps[mt][v];
      if (mr < 8) {
        sc += bn1b0;
        const int ee = e0 + mt*16 + 8*h + mr;
        const int r2 = eidx[ee], c2 = eidx[NEDGE + ee];
        const float* nr = node_attr + (size_t)r2*NAW;
        const float* nc = node_attr + (size_t)c2*NAW;
        atomicAdd(&seg_vec[(size_t)c2*3 + 0], sc*(nr[0]-nc[0]));
        atomicAdd(&seg_vec[(size_t)c2*3 + 1], sc*(nr[1]-nc[1]));
        atomicAdd(&seg_vec[(size_t)c2*3 + 2], sc*(nr[2]-nc[2]));
        atomicAdd(&seg_cnt[c2], 1.0f);
      }
    }
  }
}

/* ------------------------------------------------------------------ */
/* Kernel 2: per-node MLP (134->128->3), segment-mean add, batch accum */
/* ------------------------------------------------------------------ */
__global__ void __launch_bounds__(128)
node_kernel(const float* __restrict__ node_attr,
            const float* __restrict__ u,
            const int*   __restrict__ batch,
            const float* __restrict__ Wn2a, const float* __restrict__ bn2a,
            const float* __restrict__ Wn2b, const float* __restrict__ bn2b,
            const float* __restrict__ seg_vec, const float* __restrict__ seg_cnt,
            float* __restrict__ out_node,   /* N*70 passthrough */
            float* __restrict__ node_sum,   /* B*73 */
            float* __restrict__ node_cnt)   /* B   */
{
  extern __shared__ __align__(16) char smem2[];
  float* Wa = (float*)smem2;            /* 134*128 */
  float* ba = Wa + N2IN*HID;            /* 128 */
  float* Wb = ba + HID;                 /* 128*3 */
  float* xs = Wb + HID*3;               /* 128*136 */
  const int tid = threadIdx.x;
  for (int i = tid; i < N2IN*HID; i += 128) Wa[i] = Wn2a[i];
  if (tid < HID) ba[tid] = bn2a[tid];
  for (int i = tid; i < HID*3; i += 128) Wb[i] = Wn2b[i];
  __syncthreads();

  const int i = blockIdx.x*128 + tid;
  if (i >= NNODE) return;
  const float* na = node_attr + (size_t)i*NAW;
  const int b = batch[i];
  float* x = xs + tid*136;
  for (int k = 0; k < 64; ++k) x[k]      = na[6 + k];
  for (int k = 0; k < 6;  ++k) x[64 + k] = na[k];
  const float* ub = u + (size_t)b*UD;
  for (int k = 0; k < 64; ++k) x[70 + k] = ub[k];

  float o0 = bn2b[0], o1 = bn2b[1], o2 = bn2b[2];
  for (int j = 0; j < HID; ++j) {
    float a = ba[j];
    for (int k = 0; k < N2IN; ++k) a += x[k] * Wa[k*HID + j];
    float s = silu(a);
    o0 += s * Wb[j*3 + 0];
    o1 += s * Wb[j*3 + 1];
    o2 += s * Wb[j*3 + 2];
  }
  float invc = __builtin_amdgcn_rcpf(fmaxf(seg_cnt[i], 1.0f));
  float a0 = o0 + seg_vec[(size_t)i*3 + 0] * invc;
  float a1 = o1 + seg_vec[(size_t)i*3 + 1] * invc;
  float a2 = o2 + seg_vec[(size_t)i*3 + 2] * invc;

  float* od = out_node + (size_t)i*NAW;
  for (int t = 0; t < NAW; ++t) od[t] = na[t];

  float* nsb = node_sum + (size_t)b*73;
  atomicAdd(&nsb[0], a0); atomicAdd(&nsb[1], a1); atomicAdd(&nsb[2], a2);
  for (int t = 0; t < NAW; ++t) atomicAdd(&nsb[3 + t], na[t]);
  atomicAdd(&node_cnt[b], 1.0f);
}

/* ------------------------------------------------------------------ */
/* Kernel 3: u_new = mlp([u | segment_mean(node_cat)], 137->128->64)   */
/* ------------------------------------------------------------------ */
__global__ void global_kernel(const float* __restrict__ u,
                              const float* __restrict__ Wga, const float* __restrict__ bga,
                              const float* __restrict__ Wgb, const float* __restrict__ bgb,
                              const float* __restrict__ node_sum, const float* __restrict__ node_cnt,
                              float* __restrict__ out_u)
{
  __shared__ float x[137];
  __shared__ float hh[128];
  const int tid = threadIdx.x;
  for (int b = 0; b < NB; ++b) {
    if (tid < 137) {
      float v;
      if (tid < 64) v = u[b*UD + tid];
      else          v = node_sum[b*73 + (tid - 64)] / fmaxf(node_cnt[b], 1.0f);
      x[tid] = v;
    }
    __syncthreads();
    if (tid < 128) {
      float a = bga[tid];
      for (int k = 0; k < 137; ++k) a += x[k] * Wga[k*HID + tid];
      hh[tid] = silu(a);
    }
    __syncthreads();
    if (tid < 64) {
      float a = bgb[tid];
      for (int j = 0; j < 128; ++j) a += hh[j] * Wgb[j*UD + tid];
      out_u[b*UD + tid] = a;
    }
    __syncthreads();
  }
}

__global__ void zero_kernel(float* __restrict__ p, int n) {
  int i = blockIdx.x*blockDim.x + threadIdx.x;
  if (i < n) p[i] = 0.0f;
}

/* ------------------------------------------------------------------ */
extern "C" void kernel_launch(void* const* d_in, const int* in_sizes, int n_in,
                              void* d_out, int out_size, void* d_ws, size_t ws_size,
                              hipStream_t stream) {
  (void)in_sizes; (void)n_in; (void)out_size; (void)ws_size;
  const float* node_attr = (const float*)d_in[0];
  const int*   eidx      = (const int*)d_in[1];
  const float* edge_attr = (const float*)d_in[2];
  const float* u         = (const float*)d_in[3];
  const int*   batch     = (const int*)d_in[4];
  const float* We1  = (const float*)d_in[5];  const float* be1  = (const float*)d_in[6];
  const float* We2  = (const float*)d_in[7];  const float* be2  = (const float*)d_in[8];
  const float* Wn1a = (const float*)d_in[9];  const float* bn1a = (const float*)d_in[10];
  const float* Wn1b = (const float*)d_in[11]; const float* bn1b = (const float*)d_in[12];
  const float* Wn2a = (const float*)d_in[13]; const float* bn2a = (const float*)d_in[14];
  const float* Wn2b = (const float*)d_in[15]; const float* bn2b = (const float*)d_in[16];
  const float* Wga  = (const float*)d_in[17]; const float* bga  = (const float*)d_in[18];
  const float* Wgb  = (const float*)d_in[19]; const float* bgb  = (const float*)d_in[20];

  float* out      = (float*)d_out;
  float* out_node = out;                                            /* N*70  */
  float* out_ef   = out + (size_t)NNODE*NAW;                        /* E*128 */
  float* out_u    = out + (size_t)NNODE*NAW + (size_t)NEDGE*HID;    /* B*64  */

  float* ws       = (float*)d_ws;
  float* seg_vec  = ws;                       /* N*3  */
  float* seg_cnt  = ws + (size_t)NNODE*3;     /* N    */
  float* node_sum = ws + (size_t)NNODE*4;     /* B*73 */
  float* node_cnt = ws + (size_t)NNODE*4 + NB*73; /* B */
  const int zn = NNODE*4 + NB*74;

  (void)hipFuncSetAttribute((const void*)edge_kernel,
      hipFuncAttributeMaxDynamicSharedMemorySize, EDGE_SMEM);
  (void)hipFuncSetAttribute((const void*)node_kernel,
      hipFuncAttributeMaxDynamicSharedMemorySize, NODE_SMEM);

  zero_kernel<<<(zn + 255)/256, 256, 0, stream>>>(ws, zn);
  edge_kernel<<<1536, 256, EDGE_SMEM, stream>>>(
      node_attr, eidx, edge_attr, We1, be1, We2, be2, Wn1a, bn1a, Wn1b, bn1b,
      out_ef, seg_vec, seg_cnt);
  node_kernel<<<(NNODE + 127)/128, 128, NODE_SMEM, stream>>>(
      node_attr, u, batch, Wn2a, bn2a, Wn2b, bn2b, seg_vec, seg_cnt,
      out_node, node_sum, node_cnt);
  global_kernel<<<1, 160, 0, stream>>>(u, Wga, bga, Wgb, bgb, node_sum, node_cnt, out_u);
}
